// AttentionLayer_37160057045029
// MI455X (gfx1250) — matrix operand order repaired and checked by reading, not compiled, not run on hardware
//
#include <hip/hip_runtime.h>
#include <hip/hip_bf16.h>
#include <math.h>

#define NH     8
#define HD     64
#define CH     512
#define LSP    4096      // 64*64 spatial positions
#define BATCH  8
#define EPS_F  1e-6f

typedef __attribute__((ext_vector_type(16))) _Float16 v16h;
typedef __attribute__((ext_vector_type(8)))  _Float16 v8h;
typedef __attribute__((ext_vector_type(8)))  float    v8f;
typedef __attribute__((ext_vector_type(4)))  unsigned int u32x4;
typedef __attribute__((ext_vector_type(8)))  int i32x8;
typedef __attribute__((ext_vector_type(4)))  int i32x4;

union F16x16 { v16h v; v8h h[2]; };

#if __has_builtin(__builtin_amdgcn_tensor_load_to_lds)
#define USE_TDM 1
#else
#define USE_TDM 0
#endif

// ---------------------------------------------------------------------------
// WMMA fragment loaders from LDS (f16, wave32, 16x16x32 shape).
// A-matrix 16x32 (MxK): m = lane%16; halves 0..7 -> K = 8*(lane>=16)+0..7,
//                       halves 8..15 -> K = 16 + 8*(lane>=16)+0..7.
// B-matrix 32x16 (KxN): n = lane%16; K same as A (8*(lane>=16)+0..7, then +16).
// C/D 16x16 f32: row m = r + 8*(lane>=16), col n = lane%16.
// ---------------------------------------------------------------------------
__device__ __forceinline__ v16h load_afrag(const _Float16* lds, int stride, int m0, int k0) {
  const int lane = threadIdx.x & 31;
  const int m  = m0 + (lane & 15);
  const int kb = k0 + ((lane >> 4) << 3);
  const _Float16* row = lds + m * stride + kb;
  F16x16 f;
  f.h[0] = *(const v8h*)(row);
  f.h[1] = *(const v8h*)(row + 16);
  return f.v;
}

__device__ __forceinline__ v16h load_bfrag(const _Float16* lds, int stride, int n0, int k0) {
  const int lane = threadIdx.x & 31;
  const int n  = n0 + (lane & 15);
  const int kb = k0 + ((lane >> 4) << 3);
  const _Float16* row = lds + n * stride + kb;
  F16x16 f;
  f.h[0] = *(const v8h*)(row);
  f.h[1] = *(const v8h*)(row + 16);
  return f.v;
}

#define WMMA_F32_F16(a, b, c) \
  __builtin_amdgcn_wmma_f32_16x16x32_f16(false, (a), false, (b), (short)0, (c), false, false)

// ---------------------------------------------------------------------------
// Tensor Data Mover: async 2D f16 tile load Global->LDS with hardware row
// padding (D# groups per CDNA5 ISA 8.3/8.4). One instruction moves the whole
// tile; completion tracked with TENSORcnt. Issue from one wave only.
//   tile_k  : elements per row (contiguous in memory)
//   rows    : number of rows (tile_dim1)
//   rstride : global row stride in elements (tensor_dim0_stride)
//   pic/pac : pad codes (interval = 2^(pic+1) DWORDs, amount = pac+1 DWORDs)
// ---------------------------------------------------------------------------
#if USE_TDM
__device__ __forceinline__ void tdm_load_2d_f16(const _Float16* gsrc, const _Float16* ldst,
                                                unsigned tile_k, unsigned rows,
                                                unsigned rstride, unsigned pic, unsigned pac) {
  const unsigned long long ga = (unsigned long long)(size_t)gsrc;
  const unsigned lds_off = (unsigned)(size_t)ldst;   // low 32 bits = LDS byte offset
  const unsigned TD = 1u << 20;                      // large tensor dims: tiles never OOB
  u32x4 g0 = { 1u,                                   // count=1 (valid user descriptor)
               lds_off,
               (unsigned)ga,
               (((unsigned)(ga >> 32)) & 0x01FFFFFFu) | (2u << 30) };   // addr[56:32] | type=2
  i32x8 g1;
  g1[0] = (int)((1u << 16) |          // data_size = 2 bytes
                (1u << 20) |          // pad_enable
                (pic << 22) |         // pad_interval code
                (pac << 25));         // pad_amount code
  g1[1] = (int)((TD & 0xFFFFu) << 16);                 // tensor_dim0[15:0] @ bit 48
  g1[2] = (int)((TD >> 16) | ((TD & 0xFFFFu) << 16));  // tensor_dim0 hi | tensor_dim1 lo
  g1[3] = (int)((TD >> 16) | (tile_k << 16));          // tensor_dim1 hi | tile_dim0
  g1[4] = (int)(rows & 0xFFFFu);                       // tile_dim1 (tile_dim2 = 0)
  g1[5] = (int)rstride;                                // tensor_dim0_stride[31:0]
  g1[6] = 0;                                           // stride0 hi, dim1 stride lo (unused)
  g1[7] = 0;
  const i32x4 gz4 = { 0, 0, 0, 0 };
  const i32x8 gz8 = { 0, 0, 0, 0, 0, 0, 0, 0 };
  __builtin_amdgcn_tensor_load_to_lds(g0, g1, gz4, gz4, gz8, 0);
}
#endif

// ---------------------------------------------------------------------------
// Stage 0 kernels: casts / transposes / zeroing
// ---------------------------------------------------------------------------
__global__ void k_cast_f16(const float* __restrict__ src, _Float16* __restrict__ dst, int n) {
  int i = blockIdx.x * blockDim.x + threadIdx.x;
  if (i < n) dst[i] = (_Float16)src[i];
}

__global__ void k_zero_f32(float* __restrict__ p, int n) {
  int i = blockIdx.x * blockDim.x + threadIdx.x;
  if (i < n) p[i] = 0.0f;
}

// x[b][c][L] f32 -> xT[b][L][c] f16, 64x64 LDS tile transpose
__global__ __launch_bounds__(256) void k_transpose_cast(const float* __restrict__ x,
                                                        _Float16* __restrict__ xt) {
  __shared__ float tile[64][65];
  const int b  = blockIdx.z;
  const int c0 = blockIdx.y * 64;
  const int l0 = blockIdx.x * 64;
  const int tx = threadIdx.x & 63;
  const int ty = threadIdx.x >> 6;   // 0..3
#pragma unroll
  for (int r = 0; r < 16; ++r) {
    int c = ty + r * 4;
    tile[c][tx] = x[((size_t)b * CH + c0 + c) * LSP + l0 + tx];
  }
  __syncthreads();
#pragma unroll
  for (int r = 0; r < 16; ++r) {
    int l = ty + r * 4;
    xt[((size_t)b * LSP + l0 + l) * CH + c0 + tx] = (_Float16)tile[tx][l];
  }
}

// ---------------------------------------------------------------------------
// Shared 128x128 GEMM core: D[m][n] = sum_k A[m][k] * Bs[n][k]
// A row-major [M][Kdim] f16, Bs row-major [N][Kdim] f16 (B pre-transposed).
// Block = 256 threads = 8 waves; wave owns a 64x32 region (4x2 WMMA tiles).
// TDM double-buffered: wave 0 DMAs tile k+1 while all waves WMMA tile k.
// At/Bt are double buffers of 2 * 128*40 halves each (pad -> LDS stride 40).
// ---------------------------------------------------------------------------
__device__ __forceinline__ void gemm128(const _Float16* __restrict__ A,
                                        const _Float16* __restrict__ Bs,
                                        int Kdim, int m0, int n0,
                                        _Float16* At, _Float16* Bt,
                                        v8f acc[4][2]) {
  const int tid = threadIdx.x;
  const int wid = tid >> 5;
  const int wm0 = (wid & 1) * 64;
  const int wn0 = (wid >> 1) * 32;
  const v8f zero = {0.f, 0.f, 0.f, 0.f, 0.f, 0.f, 0.f, 0.f};
#pragma unroll
  for (int mi = 0; mi < 4; ++mi)
#pragma unroll
    for (int ni = 0; ni < 2; ++ni) acc[mi][ni] = zero;

  const int BUF = 128 * 40;

#if USE_TDM
  if (wid == 0) {  // prologue: DMA tile 0 into buffer 0 (2 TDM ops in flight)
    tdm_load_2d_f16(A  + (size_t)m0 * Kdim, At, 32, 128, (unsigned)Kdim, 3, 3);
    tdm_load_2d_f16(Bs + (size_t)n0 * Kdim, Bt, 32, 128, (unsigned)Kdim, 3, 3);
  }
#endif

  for (int k0 = 0; k0 < Kdim; k0 += 32) {
    const int cur = (k0 >> 5) & 1;
    const _Float16* Ac = At + cur * BUF;
    const _Float16* Bc = Bt + cur * BUF;

#if USE_TDM
    if (wid == 0) {
      if (k0 + 32 < Kdim) {   // DMA next tile into the other buffer
        const int nxt = cur ^ 1;
        tdm_load_2d_f16(A  + (size_t)m0 * Kdim + k0 + 32, At + nxt * BUF,
                        32, 128, (unsigned)Kdim, 3, 3);
        tdm_load_2d_f16(Bs + (size_t)n0 * Kdim + k0 + 32, Bt + nxt * BUF,
                        32, 128, (unsigned)Kdim, 3, 3);
        __builtin_amdgcn_s_wait_tensorcnt(2);   // current tile's 2 DMAs retired
      } else {
        __builtin_amdgcn_s_wait_tensorcnt(0);
      }
    }
#else
    // fallback: cooperative 128x32 tile loads (rows contiguous in k)
    _Float16* Aw = (_Float16*)Ac;
    _Float16* Bw = (_Float16*)Bc;
#pragma unroll
    for (int i = tid; i < 512; i += 256) {
      const int r  = i >> 2;
      const int c8 = (i & 3) << 3;
      *(v8h*)(Aw + r * 40 + c8) = *(const v8h*)(A  + (size_t)(m0 + r) * Kdim + k0 + c8);
      *(v8h*)(Bw + r * 40 + c8) = *(const v8h*)(Bs + (size_t)(n0 + r) * Kdim + k0 + c8);
    }
#endif
    __syncthreads();
    v16h af[4];
#pragma unroll
    for (int mi = 0; mi < 4; ++mi) af[mi] = load_afrag(Ac, 40, wm0 + mi * 16, 0);
#pragma unroll
    for (int ni = 0; ni < 2; ++ni) {
      v16h bf = load_bfrag(Bc, 40, wn0 + ni * 16, 0);
#pragma unroll
      for (int mi = 0; mi < 4; ++mi)
        acc[mi][ni] = WMMA_F32_F16(af[mi], bf, acc[mi][ni]);
    }
    __syncthreads();
  }
}

// ---------------------------------------------------------------------------
// Stage 1: fused Q/K/V projection. grid.z = batch*3 + proj.
// Outputs stored [b][h][L][hd] f16 (lane rows contiguous -> b128 stores).
// ---------------------------------------------------------------------------
__global__ __launch_bounds__(256) void k_proj_qkv(
    const _Float16* __restrict__ xbT,
    const _Float16* __restrict__ wqh, const _Float16* __restrict__ wkh,
    const _Float16* __restrict__ wvh,
    const float* __restrict__ bq, const float* __restrict__ bk,
    const float* __restrict__ bv,
    _Float16* __restrict__ QT, _Float16* __restrict__ KT, _Float16* __restrict__ VT) {
  __shared__ _Float16 At[2 * 128 * 40];
  __shared__ _Float16 Bt[2 * 128 * 40];
  const int z = blockIdx.z;
  const int b = z / 3, p = z % 3;
  const _Float16* W    = (p == 0) ? wqh : (p == 1) ? wkh : wvh;
  const float*    bias = (p == 0) ? bq  : (p == 1) ? bk  : bv;
  _Float16*       outp = (p == 0) ? QT  : (p == 1) ? KT  : VT;
  const int m0 = blockIdx.y * 128;
  const int n0 = blockIdx.x * 128;

  v8f acc[4][2];
  gemm128(W, xbT + (size_t)b * LSP * CH, CH, m0, n0, At, Bt, acc);

  const int tid = threadIdx.x, lane = tid & 31, wid = tid >> 5;
  const int wm0 = (wid & 1) * 64, wn0 = (wid >> 1) * 32;
  const float invL = 1.0f / (float)LSP;
#pragma unroll
  for (int mi = 0; mi < 4; ++mi)
#pragma unroll
    for (int ni = 0; ni < 2; ++ni) {
      const int o0 = m0 + wm0 + mi * 16 + ((lane >> 4) << 3);   // 8 consecutive channels
      const int l  = n0 + wn0 + ni * 16 + (lane & 15);
      v8h ov;
#pragma unroll
      for (int r = 0; r < 8; ++r) {
        float t = acc[mi][ni][r] + bias[o0 + r];
        if (p < 2) t = (t > 0.0f) ? (t + 1.0f) : __expf(t);  // elu(x)+1
        else       t *= invL;                                // V / L
        ov[r] = (_Float16)t;
      }
      const int hh = o0 >> 6, dd = o0 & 63;
      *(v8h*)(outp + ((((size_t)b * NH + hh) * LSP + l) << 6) + dd) = ov;
    }
}

// ---------------------------------------------------------------------------
// Stage 2: KV[d][e] = sum_l K[l][d]*V[l][e];  Ksum[d] = sum_l K[l][d]
// grid (8 L-chunks, 64 bh); fp32 atomic accumulation into workspace.
// Tiles transposed into LDS [d][l] / [e][l] so both frags read contiguous.
// ---------------------------------------------------------------------------
__global__ __launch_bounds__(256) void k_kv_ksum(
    const _Float16* __restrict__ KT, const _Float16* __restrict__ VT,
    float* __restrict__ KV, float* __restrict__ Ksum) {
  __shared__ _Float16 Kt[64 * 40];
  __shared__ _Float16 Vt[64 * 40];
  __shared__ float kpart[256];
  const int bh    = blockIdx.y;
  const int chunk = blockIdx.x;
  const int tid = threadIdx.x, lane = tid & 31, wid = tid >> 5;
  const int d0 = (wid & 3) * 16;
  const int e0 = (wid >> 2) * 32;
  const size_t base = (size_t)bh * LSP * HD;

  const v8f zero = {0.f, 0.f, 0.f, 0.f, 0.f, 0.f, 0.f, 0.f};
  v8f acc0 = zero, acc1 = zero;
  float ks = 0.0f;
  const int srow = tid >> 3;        // l within 32-tile
  const int scol = (tid & 7) << 3;  // d/e base

  for (int it = 0; it < 16; ++it) {
    const int l0 = chunk * 512 + it * 32;
    // transpose-stage [l][hd] -> LDS [hd][l]
    const v8h kk = *(const v8h*)(KT + base + (size_t)(l0 + srow) * HD + scol);
    const v8h vv = *(const v8h*)(VT + base + (size_t)(l0 + srow) * HD + scol);
#pragma unroll
    for (int j = 0; j < 8; ++j) {
      Kt[(scol + j) * 40 + srow] = kk[j];
      Vt[(scol + j) * 40 + srow] = vv[j];
    }
    __syncthreads();
    {
      const int d = tid >> 2, cb = (tid & 3) << 3;
#pragma unroll
      for (int j = 0; j < 8; ++j) ks += (float)Kt[d * 40 + cb + j];
    }
    v16h a  = load_afrag(Kt, 40, d0, 0);
    v16h b0 = load_bfrag(Vt, 40, e0, 0);
    v16h b1 = load_bfrag(Vt, 40, e0 + 16, 0);
    acc0 = WMMA_F32_F16(a, b0, acc0);
    acc1 = WMMA_F32_F16(a, b1, acc1);
    __syncthreads();
  }

  const int dd = d0 + ((lane >> 4) << 3);
  const int ee = lane & 15;
#pragma unroll
  for (int r = 0; r < 8; ++r) {
    atomicAdd(KV + ((size_t)bh * HD + dd + r) * HD + e0 + ee,      acc0[r]);
    atomicAdd(KV + ((size_t)bh * HD + dd + r) * HD + e0 + 16 + ee, acc1[r]);
  }
  kpart[tid] = ks;
  __syncthreads();
  if (tid < 64) {
    float s = kpart[tid * 4] + kpart[tid * 4 + 1] + kpart[tid * 4 + 2] + kpart[tid * 4 + 3];
    atomicAdd(Ksum + bh * HD + tid, s);
  }
}

// ---------------------------------------------------------------------------
// Stage 3: attn[e][l] = (KV^T Q)[e][l] * L/(Q[l]·Ksum + eps)
// grid (32 l-tiles, 64 bh); output [b][L][C] f16 ready for final GEMM.
// Q tile is TDM-DMAed into padded LDS (stride 72) while KVT is staged by VALU.
// ---------------------------------------------------------------------------
__global__ __launch_bounds__(256) void k_attn_out(
    const _Float16* __restrict__ QT, const float* __restrict__ KV,
    const float* __restrict__ Ksum, _Float16* __restrict__ attnT) {
  __shared__ _Float16 KVT[64 * 72];   // [e][d]
  __shared__ _Float16 Qt[128 * 72];   // [l][d], stride 72 via TDM pad
  __shared__ float ksums[64];
  __shared__ float zl[128];
  const int bh = blockIdx.y, b = bh >> 3, h = bh & 7;
  const int l0 = blockIdx.x * 128;
  const int tid = threadIdx.x, lane = tid & 31, wid = tid >> 5;

#if USE_TDM
  if (wid == 0)   // rows of 64 f16 = 32 DWORDs; pad 4 DWORDs each -> stride 72
    tdm_load_2d_f16(QT + ((size_t)bh * LSP + l0) * HD, Qt, 64, 128, HD, 4, 3);
#endif

  {  // stage KV (fp32 [d][e]) transposed into f16 LDS [e][d]
    const int d = tid >> 2, eb = (tid & 3) << 4;
    const float* src = KV + ((size_t)bh * HD + d) * HD + eb;
#pragma unroll
    for (int j = 0; j < 16; ++j) KVT[(eb + j) * 72 + d] = (_Float16)src[j];
  }
  if (tid < 64) ksums[tid] = Ksum[bh * HD + tid];

#if USE_TDM
  if (wid == 0) __builtin_amdgcn_s_wait_tensorcnt(0);
#else
#pragma unroll
  for (int i = tid; i < 1024; i += 256) {   // Q tile [128 l][64 d], direct copy
    const int l = i >> 3, cb = (i & 7) << 3;
    *(v8h*)(Qt + l * 72 + cb) = *(const v8h*)(QT + ((size_t)bh * LSP + l0 + l) * HD + cb);
  }
#endif
  __syncthreads();
  if (tid < 128) {   // Z scale per position
    float s = 0.0f;
#pragma unroll
    for (int d = 0; d < 64; ++d) s += ksums[d] * (float)Qt[tid * 72 + d];
    zl[tid] = (float)LSP / (s + EPS_F);
  }
  __syncthreads();

  const int e0  = (wid & 3) * 16;
  const int ln0 = (wid >> 2) * 64;
  const v8f zero = {0.f, 0.f, 0.f, 0.f, 0.f, 0.f, 0.f, 0.f};
  v8f acc[4] = {zero, zero, zero, zero};
#pragma unroll
  for (int k0 = 0; k0 < 64; k0 += 32) {
    v16h a = load_afrag(KVT, 72, e0, k0);
#pragma unroll
    for (int ni = 0; ni < 4; ++ni) {
      v16h bb = load_bfrag(Qt, 72, ln0 + ni * 16, k0);
      acc[ni] = WMMA_F32_F16(a, bb, acc[ni]);
    }
  }
#pragma unroll
  for (int ni = 0; ni < 4; ++ni) {
    const int nn = ln0 + ni * 16 + (lane & 15);
    const int eb = e0 + ((lane >> 4) << 3);
    const float z = zl[nn];
    v8h ov;
#pragma unroll
    for (int r = 0; r < 8; ++r) ov[r] = (_Float16)(acc[ni][r] * z);
    *(v8h*)(attnT + ((size_t)b * LSP + l0 + nn) * CH + (h << 6) + eb) = ov;
  }
}

// ---------------------------------------------------------------------------
// Stage 4: output projection, fp32 result + bias
// ---------------------------------------------------------------------------
__global__ __launch_bounds__(256) void k_out_proj(
    const _Float16* __restrict__ attnT, const _Float16* __restrict__ woh,
    const float* __restrict__ bo, float* __restrict__ out) {
  __shared__ _Float16 At[2 * 128 * 40];
  __shared__ _Float16 Bt[2 * 128 * 40];
  const int b  = blockIdx.z;
  const int m0 = blockIdx.y * 128;
  const int n0 = blockIdx.x * 128;

  v8f acc[4][2];
  gemm128(woh, attnT + (size_t)b * LSP * CH, CH, m0, n0, At, Bt, acc);

  const int tid = threadIdx.x, lane = tid & 31, wid = tid >> 5;
  const int wm0 = (wid & 1) * 64, wn0 = (wid >> 1) * 32;
#pragma unroll
  for (int mi = 0; mi < 4; ++mi)
#pragma unroll
    for (int ni = 0; ni < 2; ++ni) {
      const int o0 = m0 + wm0 + mi * 16 + ((lane >> 4) << 3);
      const int l  = n0 + wn0 + ni * 16 + (lane & 15);
#pragma unroll
      for (int r = 0; r < 8; ++r)
        out[((size_t)b * CH + o0 + r) * LSP + l] = acc[mi][ni][r] + bo[o0 + r];
    }
}

// ---------------------------------------------------------------------------
extern "C" void kernel_launch(void* const* d_in, const int* in_sizes, int n_in,
                              void* d_out, int out_size, void* d_ws, size_t ws_size,
                              hipStream_t stream) {
  (void)in_sizes; (void)n_in; (void)out_size; (void)ws_size;
  const float* x  = (const float*)d_in[0];
  const float* wq = (const float*)d_in[1];
  const float* bq = (const float*)d_in[2];
  const float* wk = (const float*)d_in[3];
  const float* bk = (const float*)d_in[4];
  const float* wv = (const float*)d_in[5];
  const float* bv = (const float*)d_in[6];
  const float* wo = (const float*)d_in[7];
  const float* bo = (const float*)d_in[8];
  float* out = (float*)d_out;

  char* ws = (char*)d_ws;
  size_t off = 0;
  auto carve = [&](size_t bytes) -> char* {
    char* p = ws + off;
    off = (off + bytes + 255) & ~(size_t)255;
    return p;
  };
  const size_t wbytes = (size_t)CH * CH * sizeof(_Float16);
  _Float16* wqh  = (_Float16*)carve(wbytes);
  _Float16* wkh  = (_Float16*)carve(wbytes);
  _Float16* wvh  = (_Float16*)carve(wbytes);
  _Float16* woh  = (_Float16*)carve(wbytes);
  const size_t xbytes = (size_t)BATCH * LSP * CH * sizeof(_Float16);
  _Float16* xbT   = (_Float16*)carve(xbytes);
  _Float16* QT    = (_Float16*)carve(xbytes);
  _Float16* KT    = (_Float16*)carve(xbytes);
  _Float16* VT    = (_Float16*)carve(xbytes);
  _Float16* attnT = (_Float16*)carve(xbytes);
  float* KV   = (float*)carve((size_t)BATCH * NH * HD * HD * sizeof(float));
  float* Ksum = (float*)carve((size_t)BATCH * NH * HD * sizeof(float));

  const int wn = CH * CH;
  k_cast_f16<<<dim3((wn + 255) / 256), 256, 0, stream>>>(wq, wqh, wn);
  k_cast_f16<<<dim3((wn + 255) / 256), 256, 0, stream>>>(wk, wkh, wn);
  k_cast_f16<<<dim3((wn + 255) / 256), 256, 0, stream>>>(wv, wvh, wn);
  k_cast_f16<<<dim3((wn + 255) / 256), 256, 0, stream>>>(wo, woh, wn);
  k_transpose_cast<<<dim3(LSP / 64, CH / 64, BATCH), 256, 0, stream>>>(x, xbT);

  const int zn = BATCH * NH * HD * HD + BATCH * NH * HD;   // KV + Ksum contiguous
  k_zero_f32<<<dim3((zn + 255) / 256), 256, 0, stream>>>(KV, zn);

  k_proj_qkv<<<dim3(LSP / 128, CH / 128, BATCH * 3), 256, 0, stream>>>(
      xbT, wqh, wkh, wvh, bq, bk, bv, QT, KT, VT);
  k_kv_ksum<<<dim3(8, BATCH * NH), 256, 0, stream>>>(KT, VT, KV, Ksum);
  k_attn_out<<<dim3(LSP / 128, BATCH * NH), 256, 0, stream>>>(QT, KV, Ksum, attnT);
  k_out_proj<<<dim3(LSP / 128, CH / 128, BATCH), 256, 0, stream>>>(attnT, woh, bo, out);
}
